// GNN_29858612642364
// MI455X (gfx1250) — matrix operand order repaired and checked by reading, not compiled, not run on hardware
//
#include <hip/hip_runtime.h>
#include <hip/hip_bf16.h>

typedef _Float16 f16;
typedef __attribute__((ext_vector_type(16))) _Float16 v16h;
typedef __attribute__((ext_vector_type(8)))  float    v8f;

#define HID      128
#define N_NODES  50000
#define N_EDGES  800000
#define EDGE_NF  16
#define N_LAYERS 4
#define LN_EPS   1e-5f

#define E_K    288   // edge GEMM1 K padded (272 -> 288)
#define E_KR   272
#define N_K    256   // node GEMM1 K
#define NA1_S  264   // node A1 LDS stride (halfs)
#define OUT_S  132   // fp32 inter-GEMM tile stride (floats)
#define A2_S   136   // f16 A2 tile stride (halfs)

__device__ __forceinline__ float silu_f(float x) {
  return x * (1.0f / (1.0f + __expf(-x)));
}

union AF { uint4 u[2]; v16h v; };
union H8 { f16 h[8]; uint4 u; };
union H4 { f16 h[4]; uint2 u; };

// Async 16B global->LDS copy (CDNA5), tracked by ASYNCcnt. lds_off is the LDS
// byte offset (low 32 bits of the generic shared pointer, per ISA aperture rules).
__device__ __forceinline__ void async_ld_b128(unsigned lds_off, const void* gaddr) {
  asm volatile("global_load_async_to_lds_b128 %0, %1, off"
               :: "v"(lds_off), "v"((unsigned long long)(uintptr_t)gaddr)
               : "memory");
}
__device__ __forceinline__ void wait_async0() {
  asm volatile("s_wait_asynccnt 0" ::: "memory");
}
__device__ __forceinline__ unsigned lds_off_of(const void* p) {
  return (unsigned)(uintptr_t)p;
}

// A fragment (16x32 f16, M x K) from LDS, per documented CDNA5 layout:
// lanes 0-15: M=lane, elems 0..7 -> K=kb..kb+7, elems 8..15 -> K=kb+16..kb+23, kb=0
// lanes 16-31: same M, kb=8
__device__ __forceinline__ v16h lds_frag_a(const f16* sA, int strideH, int rowBase,
                                           int kBase, int lane) {
  const int m  = rowBase + (lane & 15);
  const int kb = (lane & 16) ? 8 : 0;
  const f16* p = sA + m * strideH + kBase + kb;
  AF f;
  f.u[0] = *(const uint4*)(p);
  f.u[1] = *(const uint4*)(p + 16);
  return f.v;
}

// B fragment (32x16 f16, K x N) built from row-major W[N,K]: B[k,n] = W[n,k].
// lanes 0-15: N=lane, K=0..7,16..23; lanes 16-31: N=lane-16, K=8..15,24..31.
__device__ __forceinline__ v16h glb_frag_b(const f16* W, int strideH, int nBase,
                                           int kBase, int lane) {
  const int n  = nBase + (lane & 15);
  const int kb = (lane & 16) ? 8 : 0;
  const f16* p = W + (long)n * strideH + kBase + kb;
  AF f;
  f.u[0] = *(const uint4*)(p);
  f.u[1] = *(const uint4*)(p + 16);
  return f.v;
}

#define WMMA_F32_F16(acc, a, b) \
  (acc) = __builtin_amdgcn_wmma_f32_16x16x32_f16(false, (a), false, (b), (short)0, (acc), false, false)

// 64-row x 128-col output tile per block; 8 waves, wave w owns columns [16w,16w+16).
template <int KTOT>
__device__ __forceinline__ void gemm_tile(const f16* sA, int aStride,
                                          const f16* __restrict__ W, int wStride,
                                          int nb, int lane, v8f acc[4]) {
  for (int ks = 0; ks < KTOT; ks += 32) {
    const v16h bfrag = glb_frag_b(W, wStride, nb, ks, lane);
#pragma unroll
    for (int rt = 0; rt < 4; rt++) {
      const v16h afrag = lds_frag_a(sA, aStride, rt * 16, ks, lane);
      WMMA_F32_F16(acc[rt], afrag, bfrag);
    }
  }
}

// Row-wise LayerNorm + SiLU over the 64x128 fp32 tile in LDS, writing f16 A2 tile.
// Wave handles 8 rows; each lane owns 4 columns; wave32 shuffle reduction.
__device__ __forceinline__ void ln_silu_to_a2(const float* o, f16* a2,
                                              const float* __restrict__ g,
                                              const float* __restrict__ b,
                                              int wave, int lane) {
  const int c = lane * 4;
  const float4 gg = *(const float4*)(g + c);
  const float4 bb = *(const float4*)(b + c);
  for (int rr = 0; rr < 8; rr++) {
    const int r = wave * 8 + rr;
    const float4 v = *(const float4*)(o + r * OUT_S + c);
    float s = v.x + v.y + v.z + v.w;
#pragma unroll
    for (int off = 16; off > 0; off >>= 1) s += __shfl_xor(s, off, 32);
    const float mu = s * (1.0f / 128.0f);
    const float dx = v.x - mu, dy = v.y - mu, dz = v.z - mu, dw = v.w - mu;
    float q = dx * dx + dy * dy + dz * dz + dw * dw;
#pragma unroll
    for (int off = 16; off > 0; off >>= 1) q += __shfl_xor(q, off, 32);
    const float rstd = rsqrtf(q * (1.0f / 128.0f) + LN_EPS);
    H4 t;
    t.h[0] = (f16)silu_f(dx * rstd * gg.x + bb.x);
    t.h[1] = (f16)silu_f(dy * rstd * gg.y + bb.y);
    t.h[2] = (f16)silu_f(dz * rstd * gg.z + bb.z);
    t.h[3] = (f16)silu_f(dw * rstd * gg.w + bb.w);
    *(uint2*)(a2 + r * A2_S + c) = t.u;
  }
}

// ---------------- utility kernels ----------------

__global__ void k_cast(const float* __restrict__ s, f16* __restrict__ d, long n) {
  long i = (long)blockIdx.x * blockDim.x + threadIdx.x;
  const long stride = (long)gridDim.x * blockDim.x;
  for (; i < n; i += stride) d[i] = (f16)s[i];
}

// fp32 [rows,kraw] -> f16 [rows,kpad], zero padded
__global__ void k_castw(const float* __restrict__ s, f16* __restrict__ d,
                        int rows, int kraw, int kpad) {
  const long n = (long)rows * kpad;
  long i = (long)blockIdx.x * blockDim.x + threadIdx.x;
  const long stride = (long)gridDim.x * blockDim.x;
  for (; i < n; i += stride) {
    const int  k = (int)(i % kpad);
    const long r = i / kpad;
    d[i] = (k < kraw) ? (f16)s[r * kraw + k] : (f16)0.0f;
  }
}

// LayerNorm over HID=128, 8 rows per 256-thread block; fp32 + f16 outputs
__global__ __launch_bounds__(256) void k_ln(const float* __restrict__ x,
                                            const float* __restrict__ g,
                                            const float* __restrict__ b,
                                            float* __restrict__ yf,
                                            f16* __restrict__ yh, int M) {
  const int r = blockIdx.x * 8 + (threadIdx.x >> 5);
  if (r >= M) return;
  const int lane = threadIdx.x & 31;
  const int c = lane * 4;
  const float4 v = *(const float4*)(x + (long)r * HID + c);
  float s = v.x + v.y + v.z + v.w;
#pragma unroll
  for (int off = 16; off > 0; off >>= 1) s += __shfl_xor(s, off, 32);
  const float mu = s * (1.0f / 128.0f);
  const float dx = v.x - mu, dy = v.y - mu, dz = v.z - mu, dw = v.w - mu;
  float q = dx * dx + dy * dy + dz * dz + dw * dw;
#pragma unroll
  for (int off = 16; off > 0; off >>= 1) q += __shfl_xor(q, off, 32);
  const float rstd = rsqrtf(q * (1.0f / 128.0f) + LN_EPS);
  const float4 gg = *(const float4*)(g + c);
  const float4 bb = *(const float4*)(b + c);
  float4 of;
  of.x = dx * rstd * gg.x + bb.x;
  of.y = dy * rstd * gg.y + bb.y;
  of.z = dz * rstd * gg.z + bb.z;
  of.w = dw * rstd * gg.w + bb.w;
  *(float4*)(yf + (long)r * HID + c) = of;
  H4 t;
  t.h[0] = (f16)of.x; t.h[1] = (f16)of.y; t.h[2] = (f16)of.z; t.h[3] = (f16)of.w;
  *(uint2*)(yh + (long)r * HID + c) = t.u;
}

// Y = act(X @ W^T + b), K=N=128. Optional fp32 and/or f16 outputs.
__global__ __launch_bounds__(256) void k_linear128(const f16* __restrict__ X,
                                                   const f16* __restrict__ W,
                                                   const float* __restrict__ bias,
                                                   float* __restrict__ yf,
                                                   f16* __restrict__ yh,
                                                   int M, int act) {
  __shared__ f16 a1[64 * A2_S];
  const int tid = threadIdx.x, lane = tid & 31, wave = tid >> 5, nb = wave * 16;
  const long m0 = (long)blockIdx.x * 64;

  __builtin_prefetch(W + (size_t)(nb + (lane & 15)) * 128, 0, 0);

  const unsigned a1b = lds_off_of(a1);
  for (int idx = tid; idx < 64 * 16; idx += 256) {
    const int r = idx >> 4;
    const int c = (idx & 15) * 8;
    const long m = m0 + r;
    if (m < M) {
      async_ld_b128(a1b + (unsigned)(r * A2_S + c) * 2u, X + m * HID + c);
    } else {
      *(uint4*)(a1 + r * A2_S + c) = make_uint4(0u, 0u, 0u, 0u);
    }
  }
  wait_async0();
  __syncthreads();

  const v8f vz = {0.f, 0.f, 0.f, 0.f, 0.f, 0.f, 0.f, 0.f};
  v8f acc[4] = {vz, vz, vz, vz};
  gemm_tile<128>(a1, A2_S, W, 128, nb, lane, acc);

  const int cg = nb + (lane & 15);
  const float bv = bias[cg];
  const int hi = (lane & 16) ? 8 : 0;
#pragma unroll
  for (int rt = 0; rt < 4; rt++) {
#pragma unroll
    for (int r = 0; r < 8; r++) {
      const long m = m0 + rt * 16 + hi + r;
      if (m < M) {
        float v = acc[rt][r] + bv;
        if (act) v = silu_f(v);
        if (yf) yf[m * HID + cg] = v;
        if (yh) yh[m * HID + cg] = (f16)v;
      }
    }
  }
}

// Fused edge MLP: async gather -> GEMM1(K=288) -> LN -> SiLU -> GEMM2(K=128) -> atomic scatter-add
__global__ __launch_bounds__(256) void k_edge(const f16* __restrict__ hn16,
                                              const int* __restrict__ erow,
                                              const int* __restrict__ ecol,
                                              const float* __restrict__ eattr,
                                              const f16* __restrict__ W1,
                                              const float* __restrict__ b1,
                                              const float* __restrict__ lg,
                                              const float* __restrict__ lb,
                                              const f16* __restrict__ W2,
                                              const float* __restrict__ b2,
                                              float* __restrict__ agg) {
  __shared__ union {
    f16 a1[64 * E_K];                                      // 36,864 B
    struct { float o[64 * OUT_S]; f16 a2[64 * A2_S]; } s;  // 51,200 B
  } sm;

  const int tid = threadIdx.x, lane = tid & 31, wave = tid >> 5, nb = wave * 16;
  const long e0 = (long)blockIdx.x * 64;   // N_EDGES % 64 == 0

  __builtin_prefetch(W1 + (size_t)(nb + (lane & 15)) * E_K, 0, 0);
  __builtin_prefetch(W2 + (size_t)(nb + (lane & 15)) * 128, 0, 0);

  // Build A1: [hn[row] | hn[col] | edge_attr | 0-pad], 36 chunks of 8 halfs per row.
  // Node-feature gathers go async (global -> LDS direct, ASYNCcnt); edge_attr chunks
  // need fp32->f16 conversion so they take the VALU path and overlap the async copies.
  const unsigned a1b = lds_off_of(sm.a1);
  for (int idx = tid; idx < 64 * 36; idx += 256) {
    const int r = idx / 36;
    const int c = (idx - r * 36) * 8;
    const long e = e0 + r;
    const unsigned dst = a1b + (unsigned)(r * E_K + c) * 2u;
    if (c < 128) {
      async_ld_b128(dst, hn16 + (long)erow[e] * HID + c);
    } else if (c < 256) {
      async_ld_b128(dst, hn16 + (long)ecol[e] * HID + (c - 128));
    } else if (c < E_KR) {
      const float* p = eattr + e * EDGE_NF + (c - 256);
      const float4 f0 = *(const float4*)(p);
      const float4 f1 = *(const float4*)(p + 4);
      H8 t;
      t.h[0] = (f16)f0.x; t.h[1] = (f16)f0.y; t.h[2] = (f16)f0.z; t.h[3] = (f16)f0.w;
      t.h[4] = (f16)f1.x; t.h[5] = (f16)f1.y; t.h[6] = (f16)f1.z; t.h[7] = (f16)f1.w;
      *(uint4*)(sm.a1 + r * E_K + c) = t.u;
    } else {
      *(uint4*)(sm.a1 + r * E_K + c) = make_uint4(0u, 0u, 0u, 0u);
    }
  }
  wait_async0();
  __syncthreads();

  const v8f vz = {0.f, 0.f, 0.f, 0.f, 0.f, 0.f, 0.f, 0.f};
  v8f acc[4] = {vz, vz, vz, vz};
  gemm_tile<E_K>(sm.a1, E_K, W1, E_K, nb, lane, acc);
  __syncthreads();   // a1 dead; reuse LDS for fp32 tile

  {
    const int cg = nb + (lane & 15);
    const float bias = b1[cg];
    const int hi = (lane & 16) ? 8 : 0;
#pragma unroll
    for (int rt = 0; rt < 4; rt++)
#pragma unroll
      for (int r = 0; r < 8; r++)
        sm.s.o[(rt * 16 + hi + r) * OUT_S + cg] = acc[rt][r] + bias;
  }
  __syncthreads();

  ln_silu_to_a2(sm.s.o, sm.s.a2, lg, lb, wave, lane);
  __syncthreads();

  v8f acc2[4] = {vz, vz, vz, vz};
  gemm_tile<128>(sm.s.a2, A2_S, W2, 128, nb, lane, acc2);

  {
    const int cg = nb + (lane & 15);
    const float bias = b2[cg];
    const int hi = (lane & 16) ? 8 : 0;
#pragma unroll
    for (int rt = 0; rt < 4; rt++) {
#pragma unroll
      for (int r = 0; r < 8; r++) {
        const long e = e0 + rt * 16 + hi + r;
        const int node = erow[e];
        atomicAdd(agg + (long)node * HID + cg, acc2[rt][r] + bias);
      }
    }
  }
}

// Fused node MLP: [hn|agg] -> GEMM1(K=256) -> LN -> SiLU -> GEMM2(K=128) -> residual
__global__ __launch_bounds__(256) void k_node(const f16* __restrict__ hn16,
                                              const float* __restrict__ hn32,
                                              const float* __restrict__ agg,
                                              const f16* __restrict__ W1,
                                              const float* __restrict__ b1,
                                              const float* __restrict__ lg,
                                              const float* __restrict__ lb,
                                              const f16* __restrict__ W2,
                                              const float* __restrict__ b2,
                                              float* __restrict__ hout) {
  __shared__ union {
    f16 a1[64 * NA1_S];                                    // 33,792 B
    struct { float o[64 * OUT_S]; f16 a2[64 * A2_S]; } s;  // 51,200 B
  } sm;

  const int tid = threadIdx.x, lane = tid & 31, wave = tid >> 5, nb = wave * 16;
  const long n0 = (long)blockIdx.x * 64;

  __builtin_prefetch(W1 + (size_t)(nb + (lane & 15)) * N_K, 0, 0);
  __builtin_prefetch(W2 + (size_t)(nb + (lane & 15)) * 128, 0, 0);

  const unsigned a1b = lds_off_of(sm.a1);
  for (int idx = tid; idx < 64 * 32; idx += 256) {
    const int r = idx >> 5;
    const int c = (idx & 31) * 8;
    const long node = n0 + r;
    if (node < N_NODES && c < 128) {
      async_ld_b128(a1b + (unsigned)(r * NA1_S + c) * 2u, hn16 + node * HID + c);
    } else if (node < N_NODES) {
      const float* p = agg + node * HID + (c - 128);
      const float4 f0 = *(const float4*)(p);
      const float4 f1 = *(const float4*)(p + 4);
      H8 t;
      t.h[0] = (f16)f0.x; t.h[1] = (f16)f0.y; t.h[2] = (f16)f0.z; t.h[3] = (f16)f0.w;
      t.h[4] = (f16)f1.x; t.h[5] = (f16)f1.y; t.h[6] = (f16)f1.z; t.h[7] = (f16)f1.w;
      *(uint4*)(sm.a1 + r * NA1_S + c) = t.u;
    } else {
      *(uint4*)(sm.a1 + r * NA1_S + c) = make_uint4(0u, 0u, 0u, 0u);
    }
  }
  wait_async0();
  __syncthreads();

  const v8f vz = {0.f, 0.f, 0.f, 0.f, 0.f, 0.f, 0.f, 0.f};
  v8f acc[4] = {vz, vz, vz, vz};
  gemm_tile<N_K>(sm.a1, NA1_S, W1, N_K, nb, lane, acc);
  __syncthreads();

  {
    const int cg = nb + (lane & 15);
    const float bias = b1[cg];
    const int hi = (lane & 16) ? 8 : 0;
#pragma unroll
    for (int rt = 0; rt < 4; rt++)
#pragma unroll
      for (int r = 0; r < 8; r++)
        sm.s.o[(rt * 16 + hi + r) * OUT_S + cg] = acc[rt][r] + bias;
  }
  __syncthreads();

  ln_silu_to_a2(sm.s.o, sm.s.a2, lg, lb, wave, lane);
  __syncthreads();

  v8f acc2[4] = {vz, vz, vz, vz};
  gemm_tile<128>(sm.s.a2, A2_S, W2, 128, nb, lane, acc2);

  {
    const int cg = nb + (lane & 15);
    const float bias = b2[cg];
    const int hi = (lane & 16) ? 8 : 0;
#pragma unroll
    for (int rt = 0; rt < 4; rt++) {
#pragma unroll
      for (int r = 0; r < 8; r++) {
        const long node = n0 + rt * 16 + hi + r;
        if (node < N_NODES)
          hout[node * HID + cg] = hn32[node * HID + cg] + acc2[rt][r] + bias;
      }
    }
  }
}

// ---------------- host orchestration ----------------

extern "C" void kernel_launch(void* const* d_in, const int* in_sizes, int n_in,
                              void* d_out, int out_size, void* d_ws, size_t ws_size,
                              hipStream_t stream) {
  const float* h_in   = (const float*)d_in[0];
  const int*   edges  = (const int*)d_in[1];
  const float* eattr  = (const float*)d_in[2];
  const float* ei_w1  = (const float*)d_in[3];
  const float* ei_b1  = (const float*)d_in[4];
  const float* ei_w2  = (const float*)d_in[5];
  const float* ei_b2  = (const float*)d_in[6];
  const float* norm_g = (const float*)d_in[7];
  const float* norm_b = (const float*)d_in[8];
  const float* e_w1   = (const float*)d_in[9];
  const float* e_b1   = (const float*)d_in[10];
  const float* e_lg   = (const float*)d_in[11];
  const float* e_lb   = (const float*)d_in[12];
  const float* e_w2   = (const float*)d_in[13];
  const float* e_b2   = (const float*)d_in[14];
  const float* n_w1   = (const float*)d_in[15];
  const float* n_b1   = (const float*)d_in[16];
  const float* n_lg   = (const float*)d_in[17];
  const float* n_lb   = (const float*)d_in[18];
  const float* n_w2   = (const float*)d_in[19];
  const float* n_b2   = (const float*)d_in[20];
  const float* eo_w1  = (const float*)d_in[21];
  const float* eo_b1  = (const float*)d_in[22];
  const float* eo_w2  = (const float*)d_in[23];
  const float* eo_b2  = (const float*)d_in[24];
  (void)in_sizes; (void)n_in; (void)out_size; (void)ws_size;

  const int* erow = edges;
  const int* ecol = edges + N_EDGES;

  char* ws = (char*)d_ws;
  size_t off = 0;
  auto carve = [&](size_t bytes) -> char* {
    char* p = ws + off;
    off += (bytes + 255) & ~(size_t)255;
    return p;
  };

  float* hcur = (float*)carve((size_t)N_NODES * HID * 4);
  float* hn32 = (float*)carve((size_t)N_NODES * HID * 4);
  f16*   hn16 = (f16*)  carve((size_t)N_NODES * HID * 2);
  float* agg  = (float*)carve((size_t)N_NODES * HID * 4);
  f16*   x16  = (f16*)  carve((size_t)N_NODES * HID * 2);
  f16* wi1 = (f16*)carve((size_t)128 * 128 * 2);
  f16* wi2 = (f16*)carve((size_t)128 * 128 * 2);
  f16* wo1 = (f16*)carve((size_t)128 * 128 * 2);
  f16* wo2 = (f16*)carve((size_t)128 * 128 * 2);
  f16* we1 = (f16*)carve((size_t)N_LAYERS * 128 * E_K * 2);
  f16* we2 = (f16*)carve((size_t)N_LAYERS * 128 * 128 * 2);
  f16* wn1 = (f16*)carve((size_t)N_LAYERS * 128 * N_K * 2);
  f16* wn2 = (f16*)carve((size_t)N_LAYERS * 128 * 128 * 2);

  const dim3 blk(256);
  const int mBlocks  = (N_NODES + 63) / 64;   // 782
  const int lnBlocks = (N_NODES + 7) / 8;     // 6250
  const int eBlocks  = N_EDGES / 64;          // 12500

  // --- weight conversion (fp32 -> f16, pad edge W1 to K=288) ---
  k_castw<<<dim3(64), blk, 0, stream>>>(ei_w1, wi1, 128, 128, 128);
  k_castw<<<dim3(64), blk, 0, stream>>>(ei_w2, wi2, 128, 128, 128);
  k_castw<<<dim3(64), blk, 0, stream>>>(eo_w1, wo1, 128, 128, 128);
  k_castw<<<dim3(64), blk, 0, stream>>>(eo_w2, wo2, 128, 128, 128);
  for (int l = 0; l < N_LAYERS; l++)
    k_castw<<<dim3(64), blk, 0, stream>>>(e_w1 + (size_t)l * 128 * E_KR,
                                          we1 + (size_t)l * 128 * E_K, 128, E_KR, E_K);
  k_castw<<<dim3(128), blk, 0, stream>>>(e_w2, we2, N_LAYERS * 128, 128, 128);
  k_castw<<<dim3(128), blk, 0, stream>>>(n_w1, wn1, N_LAYERS * 128, N_K, N_K);
  k_castw<<<dim3(128), blk, 0, stream>>>(n_w2, wn2, N_LAYERS * 128, 128, 128);

  // --- embedding_in: silu(h @ W1^T + b1) @ W2^T + b2 ---
  k_cast<<<dim3(1024), blk, 0, stream>>>(h_in, x16, (long)N_NODES * HID);
  k_linear128<<<dim3(mBlocks), blk, 0, stream>>>(x16, wi1, ei_b1, nullptr, hn16, N_NODES, 1);
  k_linear128<<<dim3(mBlocks), blk, 0, stream>>>(hn16, wi2, ei_b2, hcur, nullptr, N_NODES, 0);

  // --- message-passing layers ---
  for (int l = 0; l < N_LAYERS; l++) {
    k_ln<<<dim3(lnBlocks), blk, 0, stream>>>(hcur, norm_g + l * HID, norm_b + l * HID,
                                             hn32, hn16, N_NODES);
    hipMemsetAsync(agg, 0, (size_t)N_NODES * HID * 4, stream);
    k_edge<<<dim3(eBlocks), blk, 0, stream>>>(hn16, erow, ecol, eattr,
                                              we1 + (size_t)l * 128 * E_K, e_b1 + l * HID,
                                              e_lg + l * HID, e_lb + l * HID,
                                              we2 + (size_t)l * 128 * 128, e_b2 + l * HID,
                                              agg);
    k_node<<<dim3(mBlocks), blk, 0, stream>>>(hn16, hn32, agg,
                                              wn1 + (size_t)l * 128 * N_K, n_b1 + l * HID,
                                              n_lg + l * HID, n_lb + l * HID,
                                              wn2 + (size_t)l * 128 * 128, n_b2 + l * HID,
                                              hcur);
  }

  // --- embedding_out ---
  k_cast<<<dim3(1024), blk, 0, stream>>>(hcur, x16, (long)N_NODES * HID);
  k_linear128<<<dim3(mBlocks), blk, 0, stream>>>(x16, wo1, eo_b1, nullptr, hn16, N_NODES, 1);
  k_linear128<<<dim3(mBlocks), blk, 0, stream>>>(hn16, wo2, eo_b2, (float*)d_out, nullptr,
                                                 N_NODES, 0);
}
